// MaskRCNN_59047210385588
// MI455X (gfx1250) — matrix operand order repaired and checked by reading, not compiled, not run on hardware
//
#include <hip/hip_runtime.h>
#include <math.h>

typedef __attribute__((ext_vector_type(16))) __bf16 v16bf;
typedef __attribute__((ext_vector_type(8)))  __bf16 v8bf;
typedef __attribute__((ext_vector_type(8)))  float  v8f;

// ---------------- problem constants ----------------
#define ATOT   144000      // anchors
#define NPAD1  262144      // pow2 >= ATOT
#define PRE    6000
#define NW     188         // ceil(6000/32)
#define POST   300
#define FC_    256
#define FH_    100
#define FW_    160
#define IMGW   2560.0f
#define IMGH   1600.0f
#define HID_   1024
#define NC_    21
#define MC_    128
#define K1_    12544       // 256*7*7
#define NDET   6000        // 300*20
#define NPAD2  8192

// ---------------- helpers ----------------
__device__ __forceinline__ unsigned int ford(float f) {
  unsigned int u = __float_as_uint(f);
  return (u & 0x80000000u) ? ~u : (u | 0x80000000u);
}
__device__ __forceinline__ float funord(unsigned int o) {
  unsigned int u = (o & 0x80000000u) ? (o ^ 0x80000000u) : ~o;
  return __uint_as_float(u);
}
__device__ __forceinline__ v8bf zero8() {
  v8bf z;
#pragma unroll
  for (int q = 0; q < 8; ++q) z[q] = (__bf16)0.0f;
  return z;
}

// ---------------- RPN: softmax score + box decode + pack sort keys ----------------
__global__ void k_rpn(const float* __restrict__ logits, const float* __restrict__ deltas,
                      const float* __restrict__ anchors, float* __restrict__ boxes,
                      unsigned long long* __restrict__ keys) {
  int i = blockIdx.x * blockDim.x + threadIdx.x;
  if (i >= NPAD1) return;
  if (i >= ATOT) { keys[i] = 0ull; return; }
  float l0 = logits[i * 2 + 0], l1 = logits[i * 2 + 1];
  float s = 1.0f / (1.0f + expf(l0 - l1));
  float ax1 = anchors[i * 4 + 0], ay1 = anchors[i * 4 + 1];
  float ax2 = anchors[i * 4 + 2], ay2 = anchors[i * 4 + 3];
  float w = ax2 - ax1, h = ay2 - ay1;
  float cx = ax1 + 0.5f * w, cy = ay1 + 0.5f * h;
  float dx = deltas[i * 4 + 0], dy = deltas[i * 4 + 1];
  float dw = deltas[i * 4 + 2], dh = deltas[i * 4 + 3];
  float pcx = dx * w + cx, pcy = dy * h + cy;
  float pw = w * expf(fminf(dw, 4.0f)), ph = h * expf(fminf(dh, 4.0f));
  float x1 = fminf(fmaxf(pcx - 0.5f * pw, 0.0f), IMGW);
  float y1 = fminf(fmaxf(pcy - 0.5f * ph, 0.0f), IMGH);
  float x2 = fminf(fmaxf(pcx + 0.5f * pw, 0.0f), IMGW);
  float y2 = fminf(fmaxf(pcy + 0.5f * ph, 0.0f), IMGH);
  boxes[i * 4 + 0] = x1; boxes[i * 4 + 1] = y1;
  boxes[i * 4 + 2] = x2; boxes[i * 4 + 3] = y2;
  bool ok = (x2 - x1 >= 16.0f) && (y2 - y1 >= 16.0f);
  float sm = ok ? s : -__builtin_inff();
  keys[i] = ((unsigned long long)ford(sm) << 32) | (unsigned long long)(0xFFFFFFFFu - (unsigned)i);
}

// ---------------- global-memory bitonic sort step (descending) ----------------
__global__ void k_bitonic(unsigned long long* __restrict__ keys, int n, int j, int k) {
  int i = blockIdx.x * blockDim.x + threadIdx.x;
  if (i >= n) return;
  int p = i ^ j;
  if (p > i) {
    unsigned long long a = keys[i], b = keys[p];
    bool up = ((i & k) == 0);
    if (up ? (a < b) : (a > b)) { keys[i] = b; keys[p] = a; }
  }
}

__global__ void k_gather_top(const unsigned long long* __restrict__ keys,
                             const float* __restrict__ boxesAll,
                             float* __restrict__ boxesTop, float* __restrict__ vals) {
  int i = blockIdx.x * blockDim.x + threadIdx.x;
  if (i >= PRE) return;
  unsigned long long key = keys[i];
  unsigned idx = 0xFFFFFFFFu - (unsigned)(key & 0xFFFFFFFFull);
  vals[i] = funord((unsigned)(key >> 32));
  for (int q = 0; q < 4; ++q) boxesTop[i * 4 + q] = boxesAll[(size_t)idx * 4 + q];
}

// ---------------- NMS: suppression bitmask + serial scan ----------------
__global__ void k_iou(const float* __restrict__ b, int n, int nw, float thresh,
                      unsigned int* __restrict__ mask) {
  int idx = blockIdx.x * blockDim.x + threadIdx.x;
  if (idx >= n * nw) return;
  int i = idx / nw, w = idx % nw;
  float ax1 = b[i * 4 + 0], ay1 = b[i * 4 + 1], ax2 = b[i * 4 + 2], ay2 = b[i * 4 + 3];
  float aa = (ax2 - ax1) * (ay2 - ay1);
  unsigned bits = 0;
  for (int q = 0; q < 32; ++q) {
    int j = w * 32 + q;
    if (j <= i || j >= n) continue;
    float bx1 = b[j * 4 + 0], by1 = b[j * 4 + 1], bx2 = b[j * 4 + 2], by2 = b[j * 4 + 3];
    float ab = (bx2 - bx1) * (by2 - by1);
    float lx = fmaxf(ax1, bx1), ly = fmaxf(ay1, by1);
    float rx = fminf(ax2, bx2), ry = fminf(ay2, by2);
    float iw = fmaxf(rx - lx, 0.0f), ih = fmaxf(ry - ly, 0.0f);
    float inter = iw * ih;
    float iou = inter / (aa + ab - inter + 1e-9f);
    if (iou > thresh) bits |= (1u << q);
  }
  mask[(size_t)i * nw + w] = bits;
}

__global__ void k_nms_scan(const unsigned int* __restrict__ mask, int* __restrict__ keep,
                           int n, int nw) {
  __shared__ unsigned int remv[192];
  __shared__ int flag;
  int t = threadIdx.x;
  for (int w = t; w < nw; w += blockDim.x) remv[w] = 0;
  __syncthreads();
  for (int i = 0; i < n; ++i) {
    if (t == 0) flag = (remv[i >> 5] >> (i & 31)) & 1;
    __syncthreads();
    int rm = flag;
    if (t == 0) keep[i] = !rm;
    if (!rm)
      for (int w = t; w < nw; w += blockDim.x) remv[w] |= mask[(size_t)i * nw + w];
    __syncthreads();
  }
}

// ---------------- select POST rois reproducing top_k tie semantics ----------------
__global__ void k_select(const float* __restrict__ vals, const int* __restrict__ keep,
                         const float* __restrict__ boxesTop, float* __restrict__ rois) {
  if (threadIdx.x != 0 || blockIdx.x != 0) return;
  int cnt = 0;
  for (int i = 0; i < PRE && cnt < POST; ++i)
    if (keep[i] && vals[i] > -1e30f) {
      for (int q = 0; q < 4; ++q) rois[cnt * 4 + q] = boxesTop[i * 4 + q];
      ++cnt;
    }
  for (int i = 0; i < PRE && cnt < POST; ++i)
    if (!(keep[i] && vals[i] > -1e30f)) {
      for (int q = 0; q < 4; ++q) rois[cnt * 4 + q] = boxesTop[i * 4 + q];
      ++cnt;
    }
}

// ---------------- ROI align (bilinear), writes bf16 [r][c][py][px] ----------------
__global__ void k_roi_align(const float* __restrict__ feat, const float* __restrict__ rois,
                            __bf16* __restrict__ outp, int P, float scale) {
  long long idx = (long long)blockIdx.x * blockDim.x + threadIdx.x;
  long long total = (long long)POST * FC_ * P * P;
  if (idx >= total) return;
  int px = (int)(idx % P);
  int py = (int)((idx / P) % P);
  int c  = (int)((idx / ((long long)P * P)) % FC_);
  int r  = (int)(idx / ((long long)P * P * FC_));
  float x1 = rois[r * 4 + 0] * scale, y1 = rois[r * 4 + 1] * scale;
  float x2 = rois[r * 4 + 2] * scale, y2 = rois[r * 4 + 3] * scale;
  float bw = (x2 - x1) / P, bh = (y2 - y1) / P;
  float sx = x1 + (px + 0.5f) * bw - 0.5f;
  float sy = y1 + (py + 0.5f) * bh - 0.5f;
  float x0f = fminf(fmaxf(floorf(sx), 0.0f), (float)(FW_ - 2));
  float y0f = fminf(fmaxf(floorf(sy), 0.0f), (float)(FH_ - 2));
  float fx = fminf(fmaxf(sx - x0f, 0.0f), 1.0f);
  float fy = fminf(fmaxf(sy - y0f, 0.0f), 1.0f);
  int x0 = (int)x0f, y0 = (int)y0f;
  const float* f = feat + ((size_t)c * FH_ + y0) * FW_ + x0;
  float v00 = f[0], v01 = f[1], v10 = f[FW_], v11 = f[FW_ + 1];
  float o = v00 * (1 - fy) * (1 - fx) + v01 * (1 - fy) * fx +
            v10 * fy * (1 - fx) + v11 * fy * fx;
  outp[idx] = (__bf16)o;
}

// ---------------- dtype conversion / weight repack ----------------
__global__ void k_cvt(const float* __restrict__ s, __bf16* __restrict__ d, long long n) {
  long long i = (long long)blockIdx.x * blockDim.x + threadIdx.x;
  if (i < n) d[i] = (__bf16)s[i];
}

// wpar[p][oc][ic] = mwd[oc][ic][1-py][1-px], p = py*2+px
__global__ void k_repack_wpar(const float* __restrict__ mwd, __bf16* __restrict__ wpar) {
  int idx = blockIdx.x * blockDim.x + threadIdx.x;
  if (idx >= 4 * MC_ * MC_) return;
  int p = idx / (MC_ * MC_);
  int rem = idx % (MC_ * MC_);
  int oc = rem / MC_, ic = rem % MC_;
  int ky = 1 - (p >> 1), kx = 1 - (p & 1);
  wpar[idx] = (__bf16)mwd[((oc * MC_ + ic) * 2 + ky) * 2 + kx];
}

// ---------------- WMMA GEMM core ----------------
constexpr int APITCH = 40;   // bf16 elements; 80B row pitch (16B aligned)
constexpr int BPITCH = 40;

__device__ __forceinline__ void wmma_step(const __bf16 (*As)[APITCH],
                                          const __bf16 (*BsT)[BPITCH],
                                          v8f acc[4], int wm, int wn) {
  int lane = threadIdx.x & 31;
  int mrow = lane & 15;
  int hi = lane >> 4;
  v16bf b;
  {
    int nn = wn * 16 + mrow;
    int ko = hi * 16;
#pragma unroll
    for (int e = 0; e < 16; ++e) b[e] = BsT[nn][ko + e];
  }
#pragma unroll
  for (int s = 0; s < 4; ++s) {
    v16bf a;
    int rb = wm * 64 + s * 16 + mrow;
#pragma unroll
    for (int v = 0; v < 8; ++v) {
      int kb = 2 * v + hi * 16;
      a[2 * v]     = As[rb][kb];
      a[2 * v + 1] = As[rb][kb + 1];
    }
    acc[s] = __builtin_amdgcn_wmma_f32_16x16x32_bf16(false, a, false, b,
                                                     (short)0, acc[s], false, false);
  }
}

// Generic GEMM: C[M,N] = act(A[M,K] @ B[K,N] + bias[N]); K % 32 == 0.
__global__ __launch_bounds__(256) void k_gemm(const __bf16* __restrict__ A,
                                              const __bf16* __restrict__ B,
                                              const float* __restrict__ bias,
                                              float* __restrict__ Cf, __bf16* __restrict__ Cb,
                                              int M, int N, int K, int relu) {
  __shared__ __bf16 As[128][APITCH];
  __shared__ __bf16 Bs[64][BPITCH];
  int t = threadIdx.x;
  int bm = blockIdx.y * 128, bn = blockIdx.x * 64;
  int wave = t >> 5, lane = t & 31;
  int wm = wave >> 2, wn = wave & 3;
  v8f acc[4];
  for (int s = 0; s < 4; ++s)
    for (int i = 0; i < 8; ++i) acc[s][i] = 0.0f;
  const v8bf zro = zero8();

  for (int k0 = 0; k0 < K; k0 += 32) {
    {
      int row = t >> 1, seg = (t & 1) * 16;
      int gr = bm + row;
      const v8bf* src = (const v8bf*)(A + (size_t)gr * K + k0 + seg);
      v8bf a0 = zro, a1 = zro;
      if (gr < M) { a0 = src[0]; a1 = src[1]; }
      *(v8bf*)&As[row][seg]     = a0;
      *(v8bf*)&As[row][seg + 8] = a1;
      if (k0 + 32 < K && gr < M) __builtin_prefetch((const __bf16*)src + 32, 0, 1);
    }
    {
      int n = t >> 2, kq = (t & 3) * 8;
      int gn = bn + n;
      v8bf tmp = zro;
      if (gn < N) {
        const __bf16* bp = B + (size_t)(k0 + kq) * N + gn;
#pragma unroll
        for (int q = 0; q < 8; ++q) tmp[q] = bp[(size_t)q * N];
      }
      *(v8bf*)&Bs[n][kq] = tmp;
    }
    __syncthreads();
    wmma_step(As, Bs, acc, wm, wn);
    __syncthreads();
  }

  int mrow = lane & 15, hi = lane >> 4;
  int gn = bn + wn * 16 + mrow;
  if (gn < N) {
    float bv = bias ? bias[gn] : 0.0f;
    for (int s = 0; s < 4; ++s)
      for (int i = 0; i < 8; ++i) {
        int gm = bm + wm * 64 + s * 16 + hi * 8 + i;
        if (gm < M) {
          float v = acc[s][i] + bv;
          if (relu) v = fmaxf(v, 0.0f);
          if (Cf) Cf[(size_t)gm * N + gn] = v;
          if (Cb) Cb[(size_t)gm * N + gn] = (__bf16)v;
        }
      }
  }
}

// conv1: implicit-GEMM 3x3 s1 p1, A = mw1[128 x 2304], B = im2col(roi14), relu -> bf16
__global__ __launch_bounds__(256) void k_conv1(const __bf16* __restrict__ Wt,
                                               const __bf16* __restrict__ roi14,
                                               const float* __restrict__ bias,
                                               __bf16* __restrict__ outp, int Ntot) {
  __shared__ __bf16 As[128][APITCH];
  __shared__ __bf16 Bs[64][BPITCH];
  const int K = 2304;
  int t = threadIdx.x;
  int bn = blockIdx.x * 64;
  int wave = t >> 5, lane = t & 31;
  int wm = wave >> 2, wn = wave & 3;
  v8f acc[4];
  for (int s = 0; s < 4; ++s)
    for (int i = 0; i < 8; ++i) acc[s][i] = 0.0f;
  const v8bf zro = zero8();

  for (int k0 = 0; k0 < K; k0 += 32) {
    {
      int row = t >> 1, seg = (t & 1) * 16;
      const v8bf* src = (const v8bf*)(Wt + (size_t)row * K + k0 + seg);
      *(v8bf*)&As[row][seg]     = src[0];
      *(v8bf*)&As[row][seg + 8] = src[1];
    }
    {
      int n = t >> 2, kq = (t & 3) * 8;
      int gn = bn + n;
      int roi = gn / 196, pix = gn % 196;
      int y = pix / 14, x = pix % 14;
      v8bf tmp = zro;
      if (gn < Ntot) {
#pragma unroll
        for (int q = 0; q < 8; ++q) {
          int k = k0 + kq + q;
          int ic = k / 9, r9 = k % 9;
          int iy = y + r9 / 3 - 1, ix = x + r9 % 3 - 1;
          if (iy >= 0 && iy < 14 && ix >= 0 && ix < 14)
            tmp[q] = roi14[((size_t)roi * FC_ + ic) * 196 + iy * 14 + ix];
        }
      }
      *(v8bf*)&Bs[n][kq] = tmp;
    }
    __syncthreads();
    wmma_step(As, Bs, acc, wm, wn);
    __syncthreads();
  }

  int mrow = lane & 15, hi = lane >> 4;
  int gn = bn + wn * 16 + mrow;
  if (gn < Ntot) {
    int roi = gn / 196, pix = gn % 196;
    for (int s = 0; s < 4; ++s)
      for (int i = 0; i < 8; ++i) {
        int gm = wm * 64 + s * 16 + hi * 8 + i;   // out channel
        float v = fmaxf(acc[s][i] + bias[gm], 0.0f);
        outp[((size_t)roi * MC_ + gm) * 196 + pix] = (__bf16)v;
      }
  }
}

// conv2: lhs-dilated 2x2 -> 4 parity GEMMs, K=128, scatter to 28x28, relu -> bf16
__global__ __launch_bounds__(256) void k_conv2(const __bf16* __restrict__ wpar,
                                               const __bf16* __restrict__ c1,
                                               const float* __restrict__ bias,
                                               __bf16* __restrict__ outp, int Ntot) {
  __shared__ __bf16 As[128][APITCH];
  __shared__ __bf16 Bs[64][BPITCH];
  const int K = 128;
  int t = threadIdx.x;
  int bn = blockIdx.x * 64;
  int par = blockIdx.z;
  int py = par >> 1, px = par & 1;
  const __bf16* A = wpar + (size_t)par * MC_ * MC_;
  int wave = t >> 5, lane = t & 31;
  int wm = wave >> 2, wn = wave & 3;
  v8f acc[4];
  for (int s = 0; s < 4; ++s)
    for (int i = 0; i < 8; ++i) acc[s][i] = 0.0f;
  const v8bf zro = zero8();

  for (int k0 = 0; k0 < K; k0 += 32) {
    {
      int row = t >> 1, seg = (t & 1) * 16;
      const v8bf* src = (const v8bf*)(A + (size_t)row * K + k0 + seg);
      *(v8bf*)&As[row][seg]     = src[0];
      *(v8bf*)&As[row][seg + 8] = src[1];
    }
    {
      int n = t >> 2, kq = (t & 3) * 8;
      int gn = bn + n;
      int roi = gn / 196, pix = gn % 196;
      v8bf tmp = zro;
      if (gn < Ntot) {
        const __bf16* bp = c1 + ((size_t)roi * MC_ + k0 + kq) * 196 + pix;
#pragma unroll
        for (int q = 0; q < 8; ++q) tmp[q] = bp[(size_t)q * 196];
      }
      *(v8bf*)&Bs[n][kq] = tmp;
    }
    __syncthreads();
    wmma_step(As, Bs, acc, wm, wn);
    __syncthreads();
  }

  int mrow = lane & 15, hi = lane >> 4;
  int gn = bn + wn * 16 + mrow;
  if (gn < Ntot) {
    int roi = gn / 196, pix = gn % 196;
    int iy = pix / 14, ix = pix % 14;
    int oy = 2 * iy + py, ox = 2 * ix + px;
    for (int s = 0; s < 4; ++s)
      for (int i = 0; i < 8; ++i) {
        int gm = wm * 64 + s * 16 + hi * 8 + i;
        float v = fmaxf(acc[s][i] + bias[gm], 0.0f);
        outp[((size_t)roi * MC_ + gm) * 784 + oy * 28 + ox] = (__bf16)v;
      }
  }
}

// conv3: 1x1, M=21, K=128 -> f32 mask logits [roi][cls][784]
__global__ __launch_bounds__(256) void k_conv3(const __bf16* __restrict__ Wo,
                                               const __bf16* __restrict__ m2,
                                               const float* __restrict__ bias,
                                               float* __restrict__ outp, int Ntot) {
  __shared__ __bf16 As[128][APITCH];
  __shared__ __bf16 Bs[64][BPITCH];
  const int K = 128;
  int t = threadIdx.x;
  int bn = blockIdx.x * 64;
  int wave = t >> 5, lane = t & 31;
  int wm = wave >> 2, wn = wave & 3;
  v8f acc[4];
  for (int s = 0; s < 4; ++s)
    for (int i = 0; i < 8; ++i) acc[s][i] = 0.0f;
  const v8bf zro = zero8();

  for (int k0 = 0; k0 < K; k0 += 32) {
    {
      int row = t >> 1, seg = (t & 1) * 16;
      const v8bf* src = (const v8bf*)(Wo + (size_t)row * K + k0 + seg);
      v8bf a0 = zro, a1 = zro;
      if (row < NC_) { a0 = src[0]; a1 = src[1]; }
      *(v8bf*)&As[row][seg]     = a0;
      *(v8bf*)&As[row][seg + 8] = a1;
    }
    {
      int n = t >> 2, kq = (t & 3) * 8;
      int gn = bn + n;
      int roi = gn / 784, pix = gn % 784;
      v8bf tmp = zro;
      if (gn < Ntot) {
        const __bf16* bp = m2 + ((size_t)roi * MC_ + k0 + kq) * 784 + pix;
#pragma unroll
        for (int q = 0; q < 8; ++q) tmp[q] = bp[(size_t)q * 784];
      }
      *(v8bf*)&Bs[n][kq] = tmp;
    }
    __syncthreads();
    wmma_step(As, Bs, acc, wm, wn);
    __syncthreads();
  }

  int mrow = lane & 15, hi = lane >> 4;
  int gn = bn + wn * 16 + mrow;
  if (gn < Ntot) {
    int roi = gn / 784, pix = gn % 784;
    for (int s = 0; s < 4; ++s)
      for (int i = 0; i < 8; ++i) {
        int gm = wm * 64 + s * 16 + hi * 8 + i;
        if (gm < NC_)
          outp[((size_t)roi * NC_ + gm) * 784 + pix] = acc[s][i] + bias[gm];
      }
  }
}

// ---------------- detection head post: softmax, decode, pack keys ----------------
__global__ void k_head(const float* __restrict__ cls, const float* __restrict__ dlt,
                       const float* __restrict__ rois, float* __restrict__ bx2,
                       unsigned long long* __restrict__ keys2) {
  int t = blockIdx.x * blockDim.x + threadIdx.x;
  if (t >= NPAD2) return;
  if (t >= NDET) { keys2[t] = 0ull; return; }
  int r = t / 20, c = t % 20 + 1;
  const float* cl = cls + r * NC_;
  float mx = cl[0];
  for (int k = 1; k < NC_; ++k) mx = fmaxf(mx, cl[k]);
  float sum = 0.0f;
  for (int k = 0; k < NC_; ++k) sum += expf(cl[k] - mx);
  float sc = expf(cl[c] - mx) / sum;

  float rx1 = rois[r * 4 + 0], ry1 = rois[r * 4 + 1];
  float rx2 = rois[r * 4 + 2], ry2 = rois[r * 4 + 3];
  float w = rx2 - rx1, h = ry2 - ry1;
  float cx = rx1 + 0.5f * w, cy = ry1 + 0.5f * h;
  const float* d = dlt + r * (NC_ * 4) + c * 4;
  float pcx = d[0] * w + cx, pcy = d[1] * h + cy;
  float pw = w * expf(fminf(d[2], 4.0f)), ph = h * expf(fminf(d[3], 4.0f));
  float x1 = fminf(fmaxf(pcx - 0.5f * pw, 0.0f), IMGW);
  float y1 = fminf(fmaxf(pcy - 0.5f * ph, 0.0f), IMGH);
  float x2 = fminf(fmaxf(pcx + 0.5f * pw, 0.0f), IMGW);
  float y2 = fminf(fmaxf(pcy + 0.5f * ph, 0.0f), IMGH);
  bx2[t * 4 + 0] = x1; bx2[t * 4 + 1] = y1;
  bx2[t * 4 + 2] = x2; bx2[t * 4 + 3] = y2;
  bool valid = (sc > 0.5f) && (x2 - x1 >= 1.0f) && (y2 - y1 >= 1.0f);
  float scm = valid ? sc : -__builtin_inff();
  keys2[t] = ((unsigned long long)ford(scm) << 32) | (unsigned long long)(0xFFFFFFFFu - (unsigned)t);
}

__global__ void k_final_gather(const unsigned long long* __restrict__ keys2,
                               const float* __restrict__ bx2,
                               float* __restrict__ dv, int* __restrict__ order,
                               float* __restrict__ bxs,
                               float* __restrict__ out_dbx, float* __restrict__ out_dci) {
  int i = blockIdx.x * blockDim.x + threadIdx.x;
  if (i >= NDET) return;
  unsigned long long key = keys2[i];
  unsigned j = 0xFFFFFFFFu - (unsigned)(key & 0xFFFFFFFFull);
  dv[i] = funord((unsigned)(key >> 32));
  order[i] = (int)j;
  for (int q = 0; q < 4; ++q) {
    float v = bx2[(size_t)j * 4 + q];
    bxs[i * 4 + q] = v;
    out_dbx[i * 4 + q] = v;
  }
  out_dci[i] = (float)(j % 20);
}

__global__ void k_dml(const int* __restrict__ order, const float* __restrict__ mask_logits,
                      float* __restrict__ out_dml) {
  long long idx = (long long)blockIdx.x * blockDim.x + threadIdx.x;
  if (idx >= (long long)NDET * 784) return;
  int i = (int)(idx / 784), p = (int)(idx % 784);
  int j = order[i];
  int roi = j / 20, cls = j % 20 + 1;
  out_dml[idx] = mask_logits[((size_t)roi * NC_ + cls) * 784 + p];
}

__global__ void k_final(const int* __restrict__ keep2, const float* __restrict__ dv,
                        float* __restrict__ out_scores, float* __restrict__ out_k2) {
  int i = blockIdx.x * blockDim.x + threadIdx.x;
  if (i >= NDET) return;
  bool k2 = keep2[i] && (dv[i] > -1e30f);
  out_scores[i] = k2 ? dv[i] : 0.0f;
  out_k2[i] = k2 ? 1.0f : 0.0f;
}

// ---------------- host orchestration ----------------
static inline int cdiv(long long a, long long b) { return (int)((a + b - 1) / b); }

static void run_bitonic(unsigned long long* keys, int n, hipStream_t stream) {
  for (int k = 2; k <= n; k <<= 1)
    for (int j = k >> 1; j >= 1; j >>= 1)
      k_bitonic<<<n / 256, 256, 0, stream>>>(keys, n, j, k);
}

extern "C" void kernel_launch(void* const* d_in, const int* in_sizes, int n_in,
                              void* d_out, int out_size, void* d_ws, size_t ws_size,
                              hipStream_t stream) {
  (void)in_sizes; (void)n_in; (void)out_size; (void)ws_size;
  const float* features = (const float*)d_in[0];
  const float* rpn_logits = (const float*)d_in[1];
  const float* rpn_deltas = (const float*)d_in[2];
  const float* anchors = (const float*)d_in[3];
  const float* W1 = (const float*)d_in[4];
  const float* b1 = (const float*)d_in[5];
  const float* Wc = (const float*)d_in[6];
  const float* bc = (const float*)d_in[7];
  const float* Wr = (const float*)d_in[8];
  const float* br = (const float*)d_in[9];
  const float* mw1 = (const float*)d_in[10];
  const float* mb1 = (const float*)d_in[11];
  const float* mwd = (const float*)d_in[12];
  const float* mbd = (const float*)d_in[13];
  const float* mwo = (const float*)d_in[14];
  const float* mbo = (const float*)d_in[15];

  // workspace bump allocator
  char* wp = (char*)d_ws;
  auto alloc = [&](size_t bytes) -> void* {
    void* r = (void*)wp;
    wp += (bytes + 255) & ~(size_t)255;
    return r;
  };
  float* boxesAll   = (float*)alloc((size_t)ATOT * 4 * 4);
  unsigned long long* keys1 = (unsigned long long*)alloc((size_t)NPAD1 * 8);
  float* boxesTop   = (float*)alloc((size_t)PRE * 4 * 4);
  float* vals       = (float*)alloc((size_t)PRE * 4);
  unsigned int* nmsmask = (unsigned int*)alloc((size_t)PRE * NW * 4);
  int* keep         = (int*)alloc((size_t)PRE * 4);
  float* rois       = (float*)alloc((size_t)POST * 4 * 4);
  __bf16* roi7_bf   = (__bf16*)alloc((size_t)POST * K1_ * 2);
  __bf16* roi14_bf  = (__bf16*)alloc((size_t)POST * FC_ * 196 * 2);
  __bf16* W1_bf     = (__bf16*)alloc((size_t)K1_ * HID_ * 2);
  __bf16* Wc_bf     = (__bf16*)alloc((size_t)HID_ * NC_ * 2);
  __bf16* Wr_bf     = (__bf16*)alloc((size_t)HID_ * NC_ * 4 * 2);
  __bf16* h_bf      = (__bf16*)alloc((size_t)POST * HID_ * 2);
  float* cls_logits = (float*)alloc((size_t)POST * NC_ * 4);
  float* roi_deltas = (float*)alloc((size_t)POST * NC_ * 4 * 4);
  __bf16* mw1_bf    = (__bf16*)alloc((size_t)MC_ * FC_ * 9 * 2);
  __bf16* wpar_bf   = (__bf16*)alloc((size_t)4 * MC_ * MC_ * 2);
  __bf16* mwo_bf    = (__bf16*)alloc((size_t)NC_ * MC_ * 2);
  __bf16* c1_bf     = (__bf16*)alloc((size_t)POST * MC_ * 196 * 2);
  __bf16* m2_bf     = (__bf16*)alloc((size_t)POST * MC_ * 784 * 2);
  float* mask_logits = (float*)alloc((size_t)POST * NC_ * 784 * 4);
  float* bx2        = (float*)alloc((size_t)NDET * 4 * 4);
  unsigned long long* keys2 = (unsigned long long*)alloc((size_t)NPAD2 * 8);
  float* dv         = (float*)alloc((size_t)NDET * 4);
  int* order        = (int*)alloc((size_t)NDET * 4);
  float* bxs        = (float*)alloc((size_t)NDET * 4 * 4);
  int* keep2        = (int*)alloc((size_t)NDET * 4);

  float* out = (float*)d_out;
  float* out_scores = out;
  float* out_dbx = out + NDET;
  float* out_dci = out + NDET + NDET * 4;
  float* out_k2  = out + NDET + NDET * 4 + NDET;
  float* out_dml = out + NDET + NDET * 4 + NDET + NDET;

  // ---- RPN proposals ----
  k_rpn<<<NPAD1 / 256, 256, 0, stream>>>(rpn_logits, rpn_deltas, anchors, boxesAll, keys1);
  run_bitonic(keys1, NPAD1, stream);
  k_gather_top<<<cdiv(PRE, 256), 256, 0, stream>>>(keys1, boxesAll, boxesTop, vals);
  k_iou<<<cdiv((long long)PRE * NW, 256), 256, 0, stream>>>(boxesTop, PRE, NW, 0.7f, nmsmask);
  k_nms_scan<<<1, 256, 0, stream>>>(nmsmask, keep, PRE, NW);
  k_select<<<1, 1, 0, stream>>>(vals, keep, boxesTop, rois);

  // ---- ROI align (bf16 outputs feed WMMA heads) ----
  k_roi_align<<<cdiv((long long)POST * FC_ * 49, 256), 256, 0, stream>>>(
      features, rois, roi7_bf, 7, 1.0f / 16.0f);
  k_roi_align<<<cdiv((long long)POST * FC_ * 196, 256), 256, 0, stream>>>(
      features, rois, roi14_bf, 14, 1.0f / 16.0f);

  // ---- weight conversions ----
  k_cvt<<<cdiv((long long)K1_ * HID_, 256), 256, 0, stream>>>(W1, W1_bf, (long long)K1_ * HID_);
  k_cvt<<<cdiv((long long)HID_ * NC_, 256), 256, 0, stream>>>(Wc, Wc_bf, (long long)HID_ * NC_);
  k_cvt<<<cdiv((long long)HID_ * NC_ * 4, 256), 256, 0, stream>>>(Wr, Wr_bf, (long long)HID_ * NC_ * 4);
  k_cvt<<<cdiv((long long)MC_ * FC_ * 9, 256), 256, 0, stream>>>(mw1, mw1_bf, (long long)MC_ * FC_ * 9);
  k_cvt<<<cdiv((long long)NC_ * MC_, 256), 256, 0, stream>>>(mwo, mwo_bf, (long long)NC_ * MC_);
  k_repack_wpar<<<cdiv(4 * MC_ * MC_, 256), 256, 0, stream>>>(mwd, wpar_bf);

  // ---- FC head: h = relu(roi7 @ W1 + b1); cls; deltas ----
  k_gemm<<<dim3(HID_ / 64, cdiv(POST, 128)), 256, 0, stream>>>(
      roi7_bf, W1_bf, b1, nullptr, h_bf, POST, HID_, K1_, 1);
  k_gemm<<<dim3(cdiv(NC_, 64), cdiv(POST, 128)), 256, 0, stream>>>(
      h_bf, Wc_bf, bc, cls_logits, nullptr, POST, NC_, HID_, 0);
  k_gemm<<<dim3(cdiv(NC_ * 4, 64), cdiv(POST, 128)), 256, 0, stream>>>(
      h_bf, Wr_bf, br, roi_deltas, nullptr, POST, NC_ * 4, HID_, 0);

  // ---- mask head ----
  const int N1 = POST * 196;       // 58800
  const int N3 = POST * 784;       // 235200
  k_conv1<<<dim3(cdiv(N1, 64)), 256, 0, stream>>>(mw1_bf, roi14_bf, mb1, c1_bf, N1);
  k_conv2<<<dim3(cdiv(N1, 64), 1, 4), 256, 0, stream>>>(wpar_bf, c1_bf, mbd, m2_bf, N1);
  k_conv3<<<dim3(cdiv(N3, 64)), 256, 0, stream>>>(mwo_bf, m2_bf, mbo, mask_logits, N3);

  // ---- detection post-processing ----
  k_head<<<NPAD2 / 256, 256, 0, stream>>>(cls_logits, roi_deltas, rois, bx2, keys2);
  run_bitonic(keys2, NPAD2, stream);
  k_final_gather<<<cdiv(NDET, 256), 256, 0, stream>>>(keys2, bx2, dv, order, bxs,
                                                      out_dbx, out_dci);
  k_dml<<<cdiv((long long)NDET * 784, 256), 256, 0, stream>>>(order, mask_logits, out_dml);
  k_iou<<<cdiv((long long)NDET * NW, 256), 256, 0, stream>>>(bxs, NDET, NW, 0.5f, nmsmask);
  k_nms_scan<<<1, 256, 0, stream>>>(nmsmask, keep2, NDET, NW);
  k_final<<<cdiv(NDET, 256), 256, 0, stream>>>(keep2, dv, out_scores, out_k2);
}
